// GFDreConv_7301444403515
// MI455X (gfx1250) — matrix operand order repaired and checked by reading, not compiled, not run on hardware
//
#include <hip/hip_runtime.h>

// ---- problem constants (from reference) ----
#define BB     8
#define CC     256
#define OO     256
#define WW_    64
#define HW     4096       // H*W
#define K9     2304       // 9*C
#define KSTEPS 72         // K9 / 32
#define ROWP   40         // padded B-tile row (bf16 elems) -> 80B, 16B aligned
#define BUFE   (128*ROWP) // one B buffer

typedef __attribute__((ext_vector_type(16))) __bf16 v16bf;
typedef __attribute__((ext_vector_type(8)))  __bf16 v8bf;
typedef __attribute__((ext_vector_type(8)))  float  v8f;

static __device__ inline v16bf cat16(v8bf lo, v8bf hi) {
  return __builtin_shufflevector(lo, hi, 0,1,2,3,4,5,6,7,8,9,10,11,12,13,14,15);
}

// ---------------------------------------------------------------------------
// Kernel 1: W_conv (256 x 2304 fp32) -> bf16, pre-swizzled into the exact
// 16x32 A-fragment lane layout (ISA 7.12.2).  L2-resident (1.18 MB).
// ---------------------------------------------------------------------------
__global__ __launch_bounds__(256) void pack_weights_bf16(
    const float* __restrict__ w, __bf16* __restrict__ wp) {
  int e = blockIdx.x * 256 + threadIdx.x;           // 0 .. 589823
  int j     = e & 7;
  int chunk = (e >> 3) & 1;
  int lane  = (e >> 4) & 31;
  int ks    = (e >> 9) % KSTEPS;
  int mt    = e / (512 * KSTEPS);
  int m     = mt * 16 + (lane & 15);
  int khalf = lane >> 4;
  int kcol  = ks * 32 + chunk * 16 + khalf * 8 + j;
  wp[e] = (__bf16)w[m * K9 + kcol];
}

// ---------------------------------------------------------------------------
// Kernel 2: fused gather + bf16 WMMA GEMM.
// WG = 128 out-channels x 128 pixels; 8 waves in 4(M) x 2(N); each wave 2x4
// 16x16 tiles.  B tile double-buffered in LDS with register prefetch.
// ---------------------------------------------------------------------------
__global__ __launch_bounds__(256) void gather_gemm_wmma(
    const float* __restrict__ feat,
    const int*   __restrict__ scx,
    const int*   __restrict__ scy,
    const __bf16* __restrict__ wp,
    float*       __restrict__ out) {

  __shared__ alignas(16) __bf16 ldsB[2 * BUFE];   // 20 KB, double buffered
  __shared__ int ldsPO[128 * 9];                  // pixel offset per (n, group)

  const int t    = threadIdx.x;
  const int lane = t & 31;
  const int wave = t >> 5;
  const int p0   = blockIdx.x * 128;
  const int mblk = blockIdx.y;
  const int b    = blockIdx.z;

  // --- pixel-offset table: group 0 = identity, groups 1..8 = gathered ---
  for (int e = t; e < 128 * 9; e += 256) {
    int n = e / 9, s = e % 9;
    int p = p0 + n;
    ldsPO[e] = (s == 0) ? p
             : ((scy[p * 8 + (s - 1)] - 1) * WW_ + (scx[p * 8 + (s - 1)] - 1));
  }

  const float* featB = feat + (size_t)b * CC * HW;

  // staging identity: 2 threads per B-tile row, 16 channels each
  const int nStage = t >> 1;
  const int kbase  = (t & 1) * 16;

  // compute identity
  const int waveM = wave >> 1, waveN = wave & 1;
  const int mt0   = mblk * 8 + waveM * 2;
  const __bf16* ap0 = wp + ((size_t)(mt0       * KSTEPS) * 32 + lane) * 16;
  const __bf16* ap1 = wp + ((size_t)((mt0 + 1) * KSTEPS) * 32 + lane) * 16;
  const int khalfB = (lane >> 4) * 8;

  v8f acc[2][4] = {};

  __syncthreads();

  // --- prefetch slab 0 into registers ---
  float v[16];
  {
    const float* pp = featB + ldsPO[nStage * 9] + (size_t)kbase * HW;
    #pragma unroll
    for (int j = 0; j < 16; ++j) v[j] = pp[(size_t)j * HW];
  }

  for (int kk = 0; kk < KSTEPS; ++kk) {
    // ---- convert + store current slab into buf[kk&1] (2x ds_store_b128) ----
    {
      __bf16* dst = ldsB + (kk & 1) * BUFE + nStage * ROWP + kbase;
      v8bf lo, hi;
      #pragma unroll
      for (int j = 0; j < 8; ++j) { lo[j] = (__bf16)v[j]; hi[j] = (__bf16)v[j + 8]; }
      *(v8bf*)dst       = lo;
      *(v8bf*)(dst + 8) = hi;
    }
    __syncthreads();

    // ---- prefetch next slab (clamped on last iter; redundant but in-bounds) ----
    {
      int kn = (kk + 1 < KSTEPS) ? kk + 1 : KSTEPS - 1;
      int g  = kn >> 3;
      int c0 = (kn & 7) * 32;
      const float* pp = featB + ldsPO[nStage * 9 + g] + (size_t)(c0 + kbase) * HW;
      #pragma unroll
      for (int j = 0; j < 16; ++j) v[j] = pp[(size_t)j * HW];
    }

    // ---- A fragments: fixed base + kk*1KB immediate ----
    v16bf af[2];
    {
      const __bf16* a0 = ap0 + (size_t)kk * 512;
      const __bf16* a1 = ap1 + (size_t)kk * 512;
      af[0] = cat16(*(const v8bf*)a0, *(const v8bf*)(a0 + 8));
      af[1] = cat16(*(const v8bf*)a1, *(const v8bf*)(a1 + 8));
    }

    // ---- B fragments from LDS + 8 WMMAs ----
    const __bf16* bbuf = ldsB + (kk & 1) * BUFE;
    #pragma unroll
    for (int nj = 0; nj < 4; ++nj) {
      int n = waveN * 64 + nj * 16 + (lane & 15);
      const __bf16* bp = bbuf + n * ROWP + khalfB;
      v16bf bf = cat16(*(const v8bf*)bp, *(const v8bf*)(bp + 16));
      acc[0][nj] = __builtin_amdgcn_wmma_f32_16x16x32_bf16(
          false, af[0], false, bf, (short)0, acc[0][nj], false, false);
      acc[1][nj] = __builtin_amdgcn_wmma_f32_16x16x32_bf16(
          false, af[1], false, bf, (short)0, acc[1][nj], false, false);
    }
  }

  // ---- store: D layout = lanes 0-15 M=r, lanes 16-31 M=r+8; N = lane%16 ----
  const int mhi = (lane >> 4) * 8;
  #pragma unroll
  for (int mi = 0; mi < 2; ++mi) {
    int obase = (mt0 + mi) * 16;
    #pragma unroll
    for (int nj = 0; nj < 4; ++nj) {
      int p = p0 + waveN * 64 + nj * 16 + (lane & 15);
      #pragma unroll
      for (int r = 0; r < 8; ++r) {
        int o = obase + r + mhi;
        out[(size_t)(b * OO + o) * HW + p] = acc[mi][nj][r];
      }
    }
  }
}

// ---------------------------------------------------------------------------
extern "C" void kernel_launch(void* const* d_in, const int* in_sizes, int n_in,
                              void* d_out, int out_size, void* d_ws, size_t ws_size,
                              hipStream_t stream) {
  const float* feat  = (const float*)d_in[0];   // (8,256,64,64) fp32
  const float* wconv = (const float*)d_in[1];   // (256,2304)   fp32
  const int*   scx   = (const int*)d_in[2];     // (64,64,8)    int32
  const int*   scy   = (const int*)d_in[3];     // (64,64,8)    int32
  float*       out   = (float*)d_out;           // (8,256,64,64) fp32
  __bf16*      wp    = (__bf16*)d_ws;           // 1.18 MB scratch

  pack_weights_bf16<<<(OO / 16) * KSTEPS * 32 * 16 / 256, 256, 0, stream>>>(wconv, wp);

  dim3 grid(HW / 128, OO / 128, BB);
  gather_gemm_wmma<<<grid, 256, 0, stream>>>(feat, scx, scy, wp, out);
}
